// STBlock_35897336660384
// MI455X (gfx1250) — compile-verified, matrix operand order checked by reading, not run on hardware
//
#include <hip/hip_runtime.h>

#define NN   50000
#define HH   256
#define EE   800000
#define ATTRN 107
#define KPAD 128

typedef __attribute__((ext_vector_type(16))) _Float16 v16h;
typedef __attribute__((ext_vector_type(8)))  float    v8f;

// ---------------------------------------------------------------------------
// Fragment loaders (CDNA5 wave32 WMMA 16x16x32 f16 layout)
// A (16xK tile, row-major source): lane<16 -> row=lane, elems 0..7 = K=kb..kb+7,
// elems 8..15 = K=kb+16..kb+23 ; lanes 16..31 hold the +8 K halves.
// B (KxN) is loaded from the row-major W ([N rows][K]) with identical packing.
// ---------------------------------------------------------------------------
__device__ inline v16h fragA_f32(const float* __restrict__ base, int ld, int kb, int lane) {
  int r  = lane & 15;
  int h8 = (lane >> 4) << 3;
  const float* p = base + (size_t)r * ld + kb + h8;
  v16h a;
#pragma unroll
  for (int i = 0; i < 8; ++i) a[i] = (_Float16)p[i];
#pragma unroll
  for (int i = 0; i < 8; ++i) a[8 + i] = (_Float16)p[16 + i];
  return a;
}

__device__ inline v16h fragA_f32_masked(const float* __restrict__ base, int ld, int kb,
                                        int lane, int kmax) {
  int r  = lane & 15;
  int h8 = (lane >> 4) << 3;
  const float* row = base + (size_t)r * ld;
  v16h a;
#pragma unroll
  for (int i = 0; i < 8; ++i) {
    int k = kb + h8 + i;
    a[i] = (k < kmax) ? (_Float16)row[k] : (_Float16)0.0f;
  }
#pragma unroll
  for (int i = 0; i < 8; ++i) {
    int k = kb + h8 + 16 + i;
    a[8 + i] = (k < kmax) ? (_Float16)row[k] : (_Float16)0.0f;
  }
  return a;
}

__device__ inline v16h fragB_f16(const _Float16* __restrict__ base, int ld, int kb, int lane) {
  int r  = lane & 15;
  int h8 = (lane >> 4) << 3;
  const _Float16* p = base + (size_t)r * ld + kb + h8;
  v16h a;
#pragma unroll
  for (int i = 0; i < 8; ++i) a[i] = p[i];
#pragma unroll
  for (int i = 0; i < 8; ++i) a[8 + i] = p[16 + i];
  return a;
}

__device__ inline void atomAddF(float* p, float v) { unsafeAtomicAdd(p, v); }

// ---------------------------------------------------------------------------
// Weight conversion kernels (run once per launch; weights then stay in L2)
// ---------------------------------------------------------------------------
__global__ void cvt_f16(const float* __restrict__ in, _Float16* __restrict__ out, int n) {
  int i = blockIdx.x * 256 + threadIdx.x;
  if (i < n) out[i] = (_Float16)in[i];
}

__global__ void cvt_pad_w1(const float* __restrict__ in, _Float16* __restrict__ out) {
  int i = blockIdx.x * 256 + threadIdx.x;       // over HH*KPAD
  if (i < HH * KPAD) {
    int r = i >> 7, k = i & (KPAD - 1);
    out[i] = (k < ATTRN) ? (_Float16)in[r * ATTRN + k] : (_Float16)0.0f;
  }
}

// ---------------------------------------------------------------------------
// Node GEMM: Hout[m][n] = sum_k X[m][k] * W[n][k]   (h = x @ W^T)
// 16 rows / block, 4 waves, each wave owns 64 columns (4 WMMA n-tiles).
// ---------------------------------------------------------------------------
__global__ __launch_bounds__(128)
void gemm_node(const float* __restrict__ X, const _Float16* __restrict__ Wh,
               float* __restrict__ Hout) {
  int lane = threadIdx.x & 31;
  int wave = threadIdx.x >> 5;
  int m0   = blockIdx.x * 16;
  const float* A = X + (size_t)m0 * HH;
  v8f z = {};
  v8f acc[4] = {z, z, z, z};
  for (int kb = 0; kb < HH; kb += 32) {
    v16h a = fragA_f32(A, HH, kb, lane);
#pragma unroll
    for (int t = 0; t < 4; ++t) {
      int n0 = wave * 64 + t * 16;
      v16h b = fragB_f16(Wh + (size_t)n0 * HH, HH, kb, lane);
      acc[t] = __builtin_amdgcn_wmma_f32_16x16x32_f16(false, a, false, b,
                                                      (short)0, acc[t], false, false);
    }
  }
  int ncol = lane & 15;
  int mb   = (lane >> 4) * 8;
#pragma unroll
  for (int t = 0; t < 4; ++t) {
    int n = wave * 64 + t * 16 + ncol;
#pragma unroll
    for (int r = 0; r < 8; ++r)
      Hout[(size_t)(m0 + mb + r) * HH + n] = acc[t][r];
  }
}

// ---------------------------------------------------------------------------
// Fused edge MLP: w[e] = relu(LN(attr[e]@W1^T + b1, g1, be1)) @ W2 + b2
// 16 edges / block; WMMA GEMM -> LDS -> LN stats -> relu-dot reduce.
// ---------------------------------------------------------------------------
__global__ __launch_bounds__(128)
void edge_mlp(const float* __restrict__ attr, const _Float16* __restrict__ W1h,
              const float* __restrict__ b1, const float* __restrict__ g1,
              const float* __restrict__ be1, const float* __restrict__ W2,
              const float* __restrict__ b2, float* __restrict__ wout) {
  __shared__ float tile[16][HH + 8];
  __shared__ float partS[16][8], partQ[16][8], partD[16][8];
  __shared__ float muS[16], rsS[16];

  int lane = threadIdx.x & 31, wave = threadIdx.x >> 5;
  int e0 = blockIdx.x * 16;
  const float* A = attr + (size_t)e0 * ATTRN;

  v8f z = {};
  v8f acc[4] = {z, z, z, z};
  for (int kb = 0; kb < KPAD; kb += 32) {
    v16h a = (kb + 32 <= ATTRN) ? fragA_f32(A, ATTRN, kb, lane)
                                : fragA_f32_masked(A, ATTRN, kb, lane, ATTRN);
#pragma unroll
    for (int t = 0; t < 4; ++t) {
      int n0 = wave * 64 + t * 16;
      v16h b = fragB_f16(W1h + (size_t)n0 * KPAD, KPAD, kb, lane);
      acc[t] = __builtin_amdgcn_wmma_f32_16x16x32_f16(false, a, false, b,
                                                      (short)0, acc[t], false, false);
    }
  }
  int ncol = lane & 15;
  int mb   = (lane >> 4) * 8;
#pragma unroll
  for (int t = 0; t < 4; ++t) {
    int n  = wave * 64 + t * 16 + ncol;
    float bn = b1[n];
#pragma unroll
    for (int r = 0; r < 8; ++r) tile[mb + r][n] = acc[t][r] + bn;
  }
  __syncthreads();

  int e = threadIdx.x >> 3, seg = threadIdx.x & 7;
  float s = 0.f, q = 0.f;
#pragma unroll 4
  for (int i = 0; i < 32; ++i) {
    float t = tile[e][seg * 32 + i];
    s += t; q += t * t;
  }
  partS[e][seg] = s; partQ[e][seg] = q;
  __syncthreads();
  if (threadIdx.x < 16) {
    float ss = 0.f, qq = 0.f;
#pragma unroll
    for (int i = 0; i < 8; ++i) { ss += partS[threadIdx.x][i]; qq += partQ[threadIdx.x][i]; }
    float mu  = ss * (1.0f / HH);
    float var = qq * (1.0f / HH) - mu * mu;
    muS[threadIdx.x] = mu;
    rsS[threadIdx.x] = rsqrtf(var + 1e-5f);
  }
  __syncthreads();
  float mu = muS[e], rs = rsS[e], d = 0.f;
#pragma unroll 4
  for (int i = 0; i < 32; ++i) {
    int c = seg * 32 + i;
    float t = (tile[e][c] - mu) * rs * g1[c] + be1[c];
    t = fmaxf(t, 0.f);
    d += t * W2[c];
  }
  partD[e][seg] = d;
  __syncthreads();
  if (threadIdx.x < 16) {
    float w = b2[0];
#pragma unroll
    for (int i = 0; i < 8; ++i) w += partD[threadIdx.x][i];
    wout[e0 + threadIdx.x] = w;
  }
}

// ---------------------------------------------------------------------------
// GCN aggregation helpers
// ---------------------------------------------------------------------------
__global__ void fill_f32(float* __restrict__ p, float v, int n) {
  int i = blockIdx.x * 256 + threadIdx.x;
  if (i < n) p[i] = v;
}

__global__ void deg_acc(const int* __restrict__ dst, const float* __restrict__ ew,
                        float* __restrict__ deg) {
  int i = blockIdx.x * 256 + threadIdx.x;
  if (i < EE) atomAddF(&deg[dst[i]], ew[i]);
}

__global__ void deg_inv(float* __restrict__ deg) {
  int i = blockIdx.x * 256 + threadIdx.x;
  if (i < NN) {
    float d = deg[i];
    deg[i] = (d > 0.f) ? rsqrtf(d) : 0.f;
  }
}

// self-loop term: agg[n][c] = dinv[n]^2 * h[n][c]
__global__ void self_loop(const float* __restrict__ h, const float* __restrict__ dinv,
                          float* __restrict__ agg) {
  int i = blockIdx.x * 256 + threadIdx.x;   // over NN*HH (exact)
  int n = i >> 8;
  float dv = dinv[n];
  agg[i] = dv * dv * h[i];
}

// wave-per-edge coalesced scatter: agg[dst] += dinv[src]*ew*dinv[dst] * h[src]
__global__ __launch_bounds__(256)
void scatter(const float* __restrict__ h, const int* __restrict__ src,
             const int* __restrict__ dst, const float* __restrict__ ew,
             const float* __restrict__ dinv, float* __restrict__ agg) {
  int e = (blockIdx.x * 256 + threadIdx.x) >> 5;   // 8 edges per block (exact)
  int lane = threadIdx.x & 31;
  int s = src[e], d = dst[e];
  float coef = dinv[s] * ew[e] * dinv[d];
  const float* hs = h + (size_t)s * HH;
  float* ad = agg + (size_t)d * HH;
#pragma unroll
  for (int i = 0; i < 8; ++i) {
    int c = lane + 32 * i;
    atomAddF(&ad[c], coef * hs[c]);
  }
}

// ---------------------------------------------------------------------------
// Residual + LayerNorm (+optional ReLU), wave-per-row, H=256 -> 8 vals/lane
// out = [relu]( (v-mu)*rsig*g + b ),  v = acc + (bias?) + xres
// ---------------------------------------------------------------------------
__global__ __launch_bounds__(256)
void post_ln(const float* __restrict__ acc, const float* __restrict__ bias,
             const float* __restrict__ xres, const float* __restrict__ g,
             const float* __restrict__ b, float* __restrict__ out, int relu) {
  int lane = threadIdx.x & 31;
  int row  = (blockIdx.x * 256 + threadIdx.x) >> 5;
  if (row >= NN) return;
  const float* ar = acc  + (size_t)row * HH;
  const float* xr = xres + (size_t)row * HH;
  float v[8];
  float s = 0.f;
#pragma unroll
  for (int i = 0; i < 8; ++i) {
    int c = lane + 32 * i;
    float t = ar[c] + xr[c];
    if (bias) t += bias[c];
    v[i] = t; s += t;
  }
#pragma unroll
  for (int o = 16; o > 0; o >>= 1) s += __shfl_xor(s, o, 32);
  float mu = s * (1.0f / HH);
  float q = 0.f;
#pragma unroll
  for (int i = 0; i < 8; ++i) { float d = v[i] - mu; q += d * d; }
#pragma unroll
  for (int o = 16; o > 0; o >>= 1) q += __shfl_xor(q, o, 32);
  float rs = rsqrtf(q * (1.0f / HH) + 1e-5f);
  float* orow = out + (size_t)row * HH;
#pragma unroll
  for (int i = 0; i < 8; ++i) {
    int c = lane + 32 * i;
    float t = (v[i] - mu) * rs * g[c] + b[c];
    if (relu) t = fmaxf(t, 0.f);
    orow[c] = t;
  }
}

// ---------------------------------------------------------------------------
extern "C" void kernel_launch(void* const* d_in, const int* in_sizes, int n_in,
                              void* d_out, int out_size, void* d_ws, size_t ws_size,
                              hipStream_t stream) {
  (void)in_sizes; (void)n_in; (void)out_size; (void)ws_size;

  const float* x    = (const float*)d_in[0];
  const int*   eib  = (const int*)d_in[1];
  const float* ewb  = (const float*)d_in[2];
  const int*   eit  = (const int*)d_in[3];
  const float* attr = (const float*)d_in[4];
  const float* Wb   = (const float*)d_in[5];
  const float* bb   = (const float*)d_in[6];
  const float* Wt   = (const float*)d_in[7];
  const float* bt   = (const float*)d_in[8];
  const float* W1   = (const float*)d_in[9];
  const float* b1   = (const float*)d_in[10];
  const float* g1   = (const float*)d_in[11];
  const float* be1  = (const float*)d_in[12];
  const float* W2   = (const float*)d_in[13];
  const float* b2   = (const float*)d_in[14];
  const float* gs   = (const float*)d_in[15];
  const float* bs   = (const float*)d_in[16];
  const float* gt   = (const float*)d_in[17];
  const float* btn  = (const float*)d_in[18];

  char* ws = (char*)d_ws;
  _Float16* WbH = (_Float16*)(ws);                 // 128 KB
  _Float16* WtH = (_Float16*)(ws + 131072);        // 128 KB
  _Float16* W1H = (_Float16*)(ws + 262144);        // 64 KB (padded K=128)
  size_t o = 327680;
  float* hbuf = (float*)(ws + o); o += (size_t)NN * HH * 4;   // 51.2 MB
  float* aggb = (float*)(ws + o); o += (size_t)NN * HH * 4;   // 51.2 MB
  float* x1   = (float*)(ws + o); o += (size_t)NN * HH * 4;   // 51.2 MB
  float* dinv = (float*)(ws + o); o += 204800;                // deg -> dinv
  float* wtmp = (float*)(ws + o);                             // E temporal weights

  const int* srcB = eib, *dstB = eib + EE;
  const int* srcT = eit, *dstT = eit + EE;

  // weight precision conversion (tiny, stays in L2)
  cvt_f16<<<(HH * HH + 255) / 256, 256, 0, stream>>>(Wb, WbH, HH * HH);
  cvt_f16<<<(HH * HH + 255) / 256, 256, 0, stream>>>(Wt, WtH, HH * HH);
  cvt_pad_w1<<<(HH * KPAD + 255) / 256, 256, 0, stream>>>(W1, W1H);

  // temporal edge-weight MLP (fused WMMA GEMM + LN + ReLU + dot)
  edge_mlp<<<EE / 16, 128, 0, stream>>>(attr, W1H, b1, g1, be1, W2, b2, wtmp);

  // ---- stage 1: BOLD GCN ----
  gemm_node<<<NN / 16, 128, 0, stream>>>(x, WbH, hbuf);
  fill_f32<<<(NN + 255) / 256, 256, 0, stream>>>(dinv, 1.0f, NN);   // self-loop weight
  deg_acc<<<(EE + 255) / 256, 256, 0, stream>>>(dstB, ewb, dinv);
  deg_inv<<<(NN + 255) / 256, 256, 0, stream>>>(dinv);
  self_loop<<<NN * HH / 256, 256, 0, stream>>>(hbuf, dinv, aggb);
  scatter<<<EE / 8, 256, 0, stream>>>(hbuf, srcB, dstB, ewb, dinv, aggb);
  post_ln<<<NN / 8, 256, 0, stream>>>(aggb, bb, x, gs, bs, x1, 1);

  // ---- stage 2: temporal GCN ----
  gemm_node<<<NN / 16, 128, 0, stream>>>(x1, WtH, hbuf);
  fill_f32<<<(NN + 255) / 256, 256, 0, stream>>>(dinv, 1.0f, NN);
  deg_acc<<<(EE + 255) / 256, 256, 0, stream>>>(dstT, wtmp, dinv);
  deg_inv<<<(NN + 255) / 256, 256, 0, stream>>>(dinv);
  self_loop<<<NN * HH / 256, 256, 0, stream>>>(hbuf, dinv, aggb);
  scatter<<<EE / 8, 256, 0, stream>>>(hbuf, srcT, dstT, wtmp, dinv, aggb);
  post_ln<<<NN / 8, 256, 0, stream>>>(aggb, bt, x1, gt, btn, hbuf, 1);  // x2 -> hbuf

  // ---- final residual LN (no relu, no bias) ----
  post_ln<<<NN / 8, 256, 0, stream>>>(hbuf, nullptr, x, gs, bs, (float*)d_out, 0);
}
